// MultiheadAttention_8358006358679
// MI455X (gfx1250) — matrix operand order repaired and checked by reading, not compiled, not run on hardware
//
#include <hip/hip_runtime.h>
#include <stdint.h>

// Problem constants (from reference): S=2048, B=4, D=1024, H=16, DH=64
#define S_LEN  2048
#define BATCH  4
#define DMODEL 1024
#define NHEADS 16
#define DHEAD  64

typedef __attribute__((ext_vector_type(16))) __bf16 v16bf;
typedef __attribute__((ext_vector_type(8)))  __bf16 bf16x8;
typedef __attribute__((ext_vector_type(8)))  float  v8f;
typedef __attribute__((ext_vector_type(4)))  unsigned int u32x4;
typedef __attribute__((ext_vector_type(4)))  int i32x4;
typedef __attribute__((ext_vector_type(8)))  int i32x8;

__device__ __forceinline__ __bf16 f2bf(float f) {
  uint32_t u = __float_as_uint(f);
  u += 0x7fffu + ((u >> 16) & 1u);           // round-to-nearest-even
  unsigned short h = (unsigned short)(u >> 16);
  return __builtin_bit_cast(__bf16, h);
}

__device__ __forceinline__ v16bf combine16(bf16x8 lo, bf16x8 hip8) {
  v16bf r;
#pragma unroll
  for (int e = 0; e < 8; e++) { r[e] = lo[e]; r[8 + e] = hip8[e]; }
  return r;
}

// ---------------------------------------------------------------- TDM
// Issue one Tensor-Data-Mover 2D tile load (global -> LDS) per CDNA5 ISA D#
// layout (CDNA5 ISA). bf16 elements (data_size=2B).
// LDS padding: 4 dwords after every 16 dwords -> row stride 32+8=40 elems,
// matching the bank-conflict-free LDS layout the compute side expects.
// Tracked with TENSORcnt; caller waits with s_wait_tensorcnt.
__device__ __forceinline__ void tdm_load_tile_bf16(uint32_t lds_off, const void* gptr,
                                                   uint32_t rows, uint32_t row_elems,
                                                   uint32_t stride_elems) {
  uint64_t ga = (uint64_t)(uintptr_t)gptr;
  u32x4 g0;
  g0.x = 1u;                                              // count=1 (valid user D#)
  g0.y = lds_off;                                         // lds_addr
  g0.z = (uint32_t)ga;                                    // global_addr[31:0]
  g0.w = (uint32_t)((ga >> 32) & 0x01FFFFFFu) | (2u << 30); // global_addr[56:32], type=2
  const uint32_t td0 = 0x7FFFFFFFu, td1 = 0x7FFFFFFFu;    // huge tensor dims: no OOB clip
  i32x8 g1;
  g1[0] = (int)((1u << 16)        // data_size = 2 bytes
              | (1u << 20)        // pad_enable
              | (3u << 22)        // pad_interval: 16 dwords
              | (3u << 25));      // pad_amount:   4 dwords
  g1[1] = (int)((td0 & 0xFFFFu) << 16);                             // tensor_dim0 lo16
  g1[2] = (int)(((td0 >> 16) & 0xFFFFu) | ((td1 & 0xFFFFu) << 16)); // td0 hi | td1 lo
  g1[3] = (int)(((td1 >> 16) & 0xFFFFu) | ((row_elems & 0xFFFFu) << 16)); // td1 hi | tile_dim0
  g1[4] = (int)(rows & 0xFFFFu);                                    // tile_dim1, tile_dim2=0
  g1[5] = (int)stride_elems;                                        // tensor_dim0_stride lo32
  g1[6] = 0;
  g1[7] = 0;
  i32x4 z4 = {0, 0, 0, 0};
#if defined(__clang_major__) && __clang_major__ >= 23
  i32x8 z8 = {0, 0, 0, 0, 0, 0, 0, 0};
  __builtin_amdgcn_tensor_load_to_lds(g0, g1, z4, z4, z8, 0);
#else
  __builtin_amdgcn_tensor_load_to_lds(g0, g1, z4, z4, 0);
#endif
}

// ---------------------------------------------------------------- converts
__global__ void cvt_f32_bf16(const float* __restrict__ in, __bf16* __restrict__ out, int n) {
  int i = blockIdx.x * blockDim.x + threadIdx.x;
  if (i < n) out[i] = f2bf(in[i]);
}

// qscale[n] = 1 / (exp(beta[n/64]) * sqrt(64))
__global__ void qscale_kernel(const float* __restrict__ beta, float* __restrict__ qs) {
  int i = blockIdx.x * blockDim.x + threadIdx.x;
  if (i < DMODEL) qs[i] = 1.0f / (__expf(beta[i >> 6]) * 8.0f);
}

// ---------------------------------------------------------------- GEMM
// C(M,N) = (A(M,K) @ B(N,K)^T + bias[n]) * colscale[n]
// A,B bf16 row-major. Block tile 128x128, BK=32, 8 waves (2x4), wave tile 64x32.
// Tiles staged by the TDM into double-buffered LDS; wave 0 drives the DMA
// pipeline (TENSORcnt is per-wave), workgroup barriers publish the data.
template <typename OutT>
__global__ __launch_bounds__(256)
void gemm_bf16_wmma(const __bf16* __restrict__ A, const __bf16* __restrict__ Bw,
                    OutT* __restrict__ C, const float* __restrict__ bias,
                    const float* __restrict__ colscale, int M, int N, int K) {
  constexpr int BM = 128, BN = 128, BK = 32, LSTR = 40; // 40-elem rows (TDM-padded)
  constexpr int TILE = BM * LSTR;                       // elems per LDS buffer
  __shared__ __bf16 As[2 * TILE];
  __shared__ __bf16 Bs[2 * TILE];
  const int tid  = threadIdx.x;
  const int lane = tid & 31, wave = tid >> 5;
  const int wm = wave >> 2, wn = wave & 3;
  const int hi = lane >> 4, ln = lane & 15;
  const int blockRow = blockIdx.y * BM, blockCol = blockIdx.x * BN;
  const int NK = K / BK;

  const uint32_t asBase = (uint32_t)(uintptr_t)As;  // generic low32 == LDS offset
  const uint32_t bsBase = (uint32_t)(uintptr_t)Bs;
  const __bf16* Atile = A  + (size_t)blockRow * K;
  const __bf16* Btile = Bw + (size_t)blockCol * K;

  v8f acc[4][2];
#pragma unroll
  for (int i = 0; i < 4; i++)
#pragma unroll
    for (int j = 0; j < 2; j++) { v8f z = {}; acc[i][j] = z; }

  if (wave == 0) {  // prologue: DMA tile 0 into buffer 0
    tdm_load_tile_bf16(asBase, Atile, BM, BK, K);
    tdm_load_tile_bf16(bsBase, Btile, BN, BK, K);
  }

  for (int kt = 0; kt < NK; kt++) {
    const int buf = kt & 1;
    if (wave == 0) {
      if (kt + 1 < NK) {  // DMA next tile into other buffer, then wait tile kt
        const uint32_t nb = (uint32_t)((buf ^ 1) * TILE * 2);
        tdm_load_tile_bf16(asBase + nb, Atile + (size_t)(kt + 1) * BK, BM, BK, K);
        tdm_load_tile_bf16(bsBase + nb, Btile + (size_t)(kt + 1) * BK, BN, BK, K);
        __builtin_amdgcn_s_wait_tensorcnt((short)2);   // tile kt's 2 loads done (in-order)
      } else {
        __builtin_amdgcn_s_wait_tensorcnt((short)0);
      }
    }
    __syncthreads();  // publish tile kt to all waves

    const __bf16* Asb = As + buf * TILE;
    const __bf16* Bsb = Bs + buf * TILE;

    // B fragments: 32x16, lane = col, elems 0..7 -> K=hi*8..hi*8+7, 8..15 -> K=hi*8+16..hi*8+23
    v16bf bfrag[2];
#pragma unroll
    for (int sn = 0; sn < 2; sn++) {
      int n = wn * 32 + sn * 16 + ln;
      const bf16x8* p = (const bf16x8*)(&Bsb[n * LSTR + hi * 8]);
      bfrag[sn] = combine16(p[0], p[2]);
    }
    // A fragments: 16x32, lane = row, elems 0..7 -> K=kb..kb+7, 8..15 -> K=kb+16..kb+23
#pragma unroll
    for (int sm = 0; sm < 4; sm++) {
      int r = wm * 64 + sm * 16 + ln;
      bf16x8 lo = *(const bf16x8*)(&Asb[r * LSTR + hi * 8]);
      bf16x8 h2 = *(const bf16x8*)(&Asb[r * LSTR + hi * 8 + 16]);
      v16bf a = combine16(lo, h2);
#pragma unroll
      for (int sn = 0; sn < 2; sn++)
        acc[sm][sn] = __builtin_amdgcn_wmma_f32_16x16x32_bf16(
            false, a, false, bfrag[sn], (short)0, acc[sm][sn], false, false);
    }
    __syncthreads();  // all waves done with buf before it is DMA-overwritten
  }

  // epilogue: C element r of lane -> row = base + r + 8*hi, col = base + (lane&15)
#pragma unroll
  for (int sm = 0; sm < 4; sm++) {
    int rowbase = blockRow + wm * 64 + sm * 16 + hi * 8;
#pragma unroll
    for (int sn = 0; sn < 2; sn++) {
      int col = blockCol + wn * 32 + sn * 16 + ln;
      float bcol = bias ? bias[col] : 0.0f;
      float scol = colscale ? colscale[col] : 1.0f;
#pragma unroll
      for (int r = 0; r < 8; r++) {
        float v = (acc[sm][sn][r] + bcol) * scol;
        size_t idx = (size_t)(rowbase + r) * N + col;
        if constexpr (sizeof(OutT) == 2) C[idx] = f2bf(v); else C[idx] = v;
      }
    }
  }
}

// ---------------------------------------------------------------- attention
// Flash attention with causal mask + zero-K/V sink column.
// Grid: (S/128, B*H). 8 waves/block, each wave owns 16 query rows.
__global__ __launch_bounds__(256)
void attn_kernel(const __bf16* __restrict__ Q, const __bf16* __restrict__ Kb,
                 const __bf16* __restrict__ Vb, __bf16* __restrict__ O) {
  constexpr int KSTR = 72, VSTR = 40, PSTR = 40;
  __shared__ __bf16 Ks[32 * KSTR];      // K tile [key][dh]
  __shared__ __bf16 Vt[64 * VSTR];      // V tile transposed [dh][key]
  __shared__ __bf16 Ps[8 * 16 * PSTR];  // per-wave probabilities [row][key]
  const int tid = threadIdx.x, lane = tid & 31, wave = tid >> 5;
  const int hi = lane >> 4, ln = lane & 15;
  const int b = blockIdx.y / NHEADS, h = blockIdx.y % NHEADS;
  const int q0b = blockIdx.x * 128;
  const int q0 = q0b + wave * 16;
  __bf16* Pw = &Ps[wave * 16 * PSTR];

  // Q fragments (A-matrix 16x32 layout), held in registers for whole kernel.
  v16bf qa[2];
  {
    const __bf16* qptr = Q + ((size_t)(q0 + ln) * BATCH + b) * DMODEL + h * DHEAD;
#pragma unroll
    for (int kk = 0; kk < 2; kk++) {
      bf16x8 lo = *(const bf16x8*)(qptr + kk * 32 + hi * 8);
      bf16x8 h2 = *(const bf16x8*)(qptr + kk * 32 + hi * 8 + 16);
      qa[kk] = combine16(lo, h2);
    }
  }

  float m[8], lsum[8];
  v8f acc[4];
#pragma unroll
  for (int r = 0; r < 8; r++) { m[r] = -1e30f; lsum[r] = 0.0f; }
#pragma unroll
  for (int v = 0; v < 4; v++) { v8f z = {}; acc[v] = z; }

  const int ntiles = (q0b + 128) / 32;  // keys needed by this block (causal)
  for (int t = 0; t < ntiles; t++) {
    const int kpos = t * 32;
    {  // cooperative K/V tile load; V stored transposed
      int row = tid >> 3, cc = tid & 7;
      size_t base = ((size_t)(kpos + row) * BATCH + b) * DMODEL + h * DHEAD + cc * 8;
      *(bf16x8*)(&Ks[row * KSTR + cc * 8]) = *(const bf16x8*)(Kb + base);
      bf16x8 vv = *(const bf16x8*)(Vb + base);
#pragma unroll
      for (int j = 0; j < 8; j++) Vt[(cc * 8 + j) * VSTR + row] = vv[j];
      if (t + 1 < ntiles) {  // hint next tile toward cache
        __builtin_prefetch(Kb + base + (size_t)32 * BATCH * DMODEL, 0, 0);
        __builtin_prefetch(Vb + base + (size_t)32 * BATCH * DMODEL, 0, 0);
      }
    }
    __syncthreads();

    if (kpos <= q0 + 15) {  // tile has at least one unmasked key for this wave
      const bool fullTile = (kpos + 31 <= q0);  // no masking needed (uniform branch)
      // scores: 16x32 = two 16x16 C tiles, K=64 -> 2 WMMAs each
      v8f sc[2];
#pragma unroll
      for (int j = 0; j < 2; j++) {
        v8f z = {}; sc[j] = z;
        int n = j * 16 + ln;
#pragma unroll
        for (int kk = 0; kk < 2; kk++) {
          const bf16x8* p = (const bf16x8*)(&Ks[n * KSTR + kk * 32 + hi * 8]);
          v16bf bfr = combine16(p[0], p[2]);
          sc[j] = __builtin_amdgcn_wmma_f32_16x16x32_bf16(
              false, qa[kk], false, bfr, (short)0, sc[j], false, false);
        }
      }
      // causal mask (partial tiles only) + online softmax (rows in 16-lane groups)
#pragma unroll
      for (int r = 0; r < 8; r++) {
        float s0 = sc[0][r], s1 = sc[1][r];
        if (!fullTile) {
          int qpos = q0 + r + hi * 8;
          if (kpos + ln > qpos)      s0 = -1e30f;
          if (kpos + 16 + ln > qpos) s1 = -1e30f;
        }
        float rmax = fmaxf(s0, s1);
#pragma unroll
        for (int d = 1; d < 16; d <<= 1) rmax = fmaxf(rmax, __shfl_xor(rmax, d, 32));
        float mnew = fmaxf(m[r], rmax);
        float p0 = __expf(s0 - mnew), p1 = __expf(s1 - mnew);
        float ps = p0 + p1;
#pragma unroll
        for (int d = 1; d < 16; d <<= 1) ps += __shfl_xor(ps, d, 32);
        float cs = __expf(m[r] - mnew);
        lsum[r] = lsum[r] * cs + ps;
        m[r] = mnew;
#pragma unroll
        for (int v = 0; v < 4; v++) acc[v][r] *= cs;
        Pw[(r + hi * 8) * PSTR + ln]      = f2bf(p0);
        Pw[(r + hi * 8) * PSTR + 16 + ln] = f2bf(p1);
      }
      // P (16x32) as A fragment via LDS re-layout
      bf16x8 plo = *(const bf16x8*)(&Pw[ln * PSTR + hi * 8]);
      bf16x8 phi = *(const bf16x8*)(&Pw[ln * PSTR + hi * 8 + 16]);
      v16bf pa = combine16(plo, phi);
      // PV: out 16x64 = four 16x16 tiles, K=32
#pragma unroll
      for (int v = 0; v < 4; v++) {
        int n = v * 16 + ln;
        const bf16x8* p = (const bf16x8*)(&Vt[n * VSTR + hi * 8]);
        v16bf bfr = combine16(p[0], p[2]);
        acc[v] = __builtin_amdgcn_wmma_f32_16x16x32_bf16(
            false, pa, false, bfr, (short)0, acc[v], false, false);
      }
    }
    __syncthreads();
  }

  // attention sink: extra key with score 0, value 0 -> denom += exp(-m)
#pragma unroll
  for (int r = 0; r < 8; r++) {
    float mnew = fmaxf(m[r], 0.0f);
    float cs = __expf(m[r] - mnew);
    float inv = 1.0f / (lsum[r] * cs + __expf(-mnew));
    size_t base = ((size_t)(q0 + r + hi * 8) * BATCH + b) * DMODEL + h * DHEAD;
#pragma unroll
    for (int v = 0; v < 4; v++)
      O[base + v * 16 + ln] = f2bf(acc[v][r] * cs * inv);
  }
}

// ---------------------------------------------------------------- launch
extern "C" void kernel_launch(void* const* d_in, const int* in_sizes, int n_in,
                              void* d_out, int out_size, void* d_ws, size_t ws_size,
                              hipStream_t stream) {
  const float* x    = (const float*)d_in[0];
  // d_in[1] = attn_mask: unused, causal+sink handled analytically
  const float* beta = (const float*)d_in[2];
  const float* Wq = (const float*)d_in[3];
  const float* bq = (const float*)d_in[4];
  const float* Wk = (const float*)d_in[5];
  const float* bk = (const float*)d_in[6];
  const float* Wv = (const float*)d_in[7];
  const float* bv = (const float*)d_in[8];
  const float* Wo = (const float*)d_in[9];
  const float* bo = (const float*)d_in[10];
  float* out = (float*)d_out;

  const int M = S_LEN * BATCH, D = DMODEL;
  char* ws = (char*)d_ws;
  size_t off = 0;
  auto alloc = [&](size_t bytes) -> char* {
    char* p = ws + off;
    off += (bytes + 255) & ~(size_t)255;
    return p;
  };
  __bf16* xb  = (__bf16*)alloc((size_t)M * D * 2);
  __bf16* wqb = (__bf16*)alloc((size_t)D * D * 2);
  __bf16* wkb = (__bf16*)alloc((size_t)D * D * 2);
  __bf16* wvb = (__bf16*)alloc((size_t)D * D * 2);
  __bf16* wob = (__bf16*)alloc((size_t)D * D * 2);
  __bf16* Qb  = (__bf16*)alloc((size_t)M * D * 2);
  __bf16* Kb  = (__bf16*)alloc((size_t)M * D * 2);
  __bf16* Vb  = (__bf16*)alloc((size_t)M * D * 2);
  __bf16* Ab  = (__bf16*)alloc((size_t)M * D * 2);
  float*  qs  = (float*)alloc((size_t)D * 4);

  cvt_f32_bf16<<<(M * D + 255) / 256, 256, 0, stream>>>(x, xb, M * D);
  cvt_f32_bf16<<<(D * D + 255) / 256, 256, 0, stream>>>(Wq, wqb, D * D);
  cvt_f32_bf16<<<(D * D + 255) / 256, 256, 0, stream>>>(Wk, wkb, D * D);
  cvt_f32_bf16<<<(D * D + 255) / 256, 256, 0, stream>>>(Wv, wvb, D * D);
  cvt_f32_bf16<<<(D * D + 255) / 256, 256, 0, stream>>>(Wo, wob, D * D);
  qscale_kernel<<<(D + 255) / 256, 256, 0, stream>>>(beta, qs);

  dim3 g(D / 128, M / 128);
  gemm_bf16_wmma<__bf16><<<g, 256, 0, stream>>>(xb, wqb, Qb, bq, qs, M, D, D);
  gemm_bf16_wmma<__bf16><<<g, 256, 0, stream>>>(xb, wkb, Kb, bk, nullptr, M, D, D);
  gemm_bf16_wmma<__bf16><<<g, 256, 0, stream>>>(xb, wvb, Vb, bv, nullptr, M, D, D);

  dim3 ga(S_LEN / 128, BATCH * NHEADS);
  attn_kernel<<<ga, 256, 0, stream>>>(Qb, Kb, Vb, Ab);

  gemm_bf16_wmma<float><<<g, 256, 0, stream>>>(Ab, wob, out, bo, nullptr, M, D, D);
}
